// StochasticLSTM_82617990906529
// MI455X (gfx1250) — matrix operand order repaired and checked by reading, not compiled, not run on hardware
//
#include <hip/hip_runtime.h>
#include <hip/hip_bf16.h>

typedef __bf16 bf16_t;
typedef bf16_t v16bf __attribute__((ext_vector_type(16)));
typedef bf16_t v8bf  __attribute__((ext_vector_type(8)));
typedef float  v8f   __attribute__((ext_vector_type(8)));
typedef int    v4i   __attribute__((ext_vector_type(4)));

#define B_  64
#define T_  256
#define D_  512
#define H_  1024
#define G4  4096   // 4*H

// --------------------------------------------------------------------------
// Async global->LDS copy (CDNA5 GLOBAL_LOAD_ASYNC_TO_LDS_B128, ASYNCcnt),
// with a synchronous fallback if the toolchain lacks the builtin.
// Builtin signature (from hipcc diagnostic): (int4 AS1*, int4 AS3*, Imm, Imm).
// --------------------------------------------------------------------------
#if __has_builtin(__builtin_amdgcn_global_load_async_to_lds_b128) && \
    __has_builtin(__builtin_amdgcn_s_wait_asynccnt)
#define USE_ASYNC_LDS 1
#else
#define USE_ASYNC_LDS 0
#endif

__device__ __forceinline__ void async_copy16(const bf16_t* g, bf16_t* l) {
#if USE_ASYNC_LDS
    __builtin_amdgcn_global_load_async_to_lds_b128(
        (__attribute__((address_space(1))) v4i*)(uintptr_t)g,
        (__attribute__((address_space(3))) v4i*)(uintptr_t)l, 0, 0);
#else
    *(v8bf*)l = *(const v8bf*)g;
#endif
}
__device__ __forceinline__ void async_wait_all() {
#if USE_ASYNC_LDS
    __builtin_amdgcn_s_wait_asynccnt(0);
#endif
}

// ---------------------------------------------------------------------------
// Concrete-dropout masks: z = (1 - sigmoid((logit(p)+logit(u))/0.1)) / (1-p)
// ---------------------------------------------------------------------------
__global__ void mask_kernel(const float* __restrict__ ux, const float* __restrict__ uh,
                            const float* __restrict__ dropout,
                            float* __restrict__ zx, float* __restrict__ zh) {
    int i = blockIdx.x * 256 + threadIdx.x;            // 0 .. 4BD+4BH-1
    float p  = dropout[0];
    float lp = __logf(p + 1e-7f) - __logf(1.0f - p + 1e-7f);
    float inv1mp = 1.0f / (1.0f - p);
    const int NX = 4 * B_ * D_;
    float u; float* dst; int j;
    if (i < NX) { u = ux[i];      dst = zx; j = i; }
    else        { u = uh[i - NX]; dst = zh; j = i - NX; }
    float lu = __logf(u + 1e-7f) - __logf(1.0f - u + 1e-7f);
    float s  = 1.0f / (1.0f + __expf(-(lp + lu) * 10.0f));   // /0.1 temperature
    dst[j] = (1.0f - s) * inv1mp;
}

// ---------------------------------------------------------------------------
// Weights -> bf16, transposed to column-major (KT[col*K + k]) so WMMA
// B-fragments are contiguous 16-element runs per lane.
// ---------------------------------------------------------------------------
__global__ void cvt_kernel(const float* __restrict__ Kx, const float* __restrict__ Kh,
                           bf16_t* __restrict__ KxT, bf16_t* __restrict__ KhT) {
    int i = blockIdx.x * 256 + threadIdx.x;            // 0 .. G4*D + G4*H - 1
    const int NX = G4 * D_;
    if (i < NX) {
        int col = i >> 9, d = i & (D_ - 1);
        KxT[i] = (bf16_t)Kx[d * G4 + col];
    } else {
        int j = i - NX;
        int col = j >> 10, u = j & (H_ - 1);
        KhT[j] = (bf16_t)Kh[u * G4 + col];
    }
}

__global__ void init_kernel(float* __restrict__ c, bf16_t* __restrict__ hm) {
    int idx = blockIdx.x * 256 + threadIdx.x;          // 0 .. 4*B*H-1
    hm[idx] = (bf16_t)0.0f;
    if (idx < B_ * H_) c[idx] = 0.0f;
}

// ---------------------------------------------------------------------------
// Input projection: XM[t,b,col] = sum_d (x[b,t,d]*zx[g,b,d]) * Kx[d,col] + bias
// All 4 waves of a block share (b, ttile, gate): the masked bf16 A tile
// (16 x 512) is built once in LDS, then each wave runs 16 WMMAs against its
// own 16-column slice of KxT.
// ---------------------------------------------------------------------------
__global__ __launch_bounds__(128)
void input_proj_kernel(const float* __restrict__ x, const float* __restrict__ zx,
                       const bf16_t* __restrict__ KxT, const float* __restrict__ bias,
                       float* __restrict__ XM) {
    __shared__ bf16_t sA[16 * 512];                    // 16 KB masked A tile

    const int tid = threadIdx.x;
    // block-uniform tile coordinates
    const int Wb       = blockIdx.x * 4;
    const int coltile0 = Wb & 255;
    const int g        = coltile0 >> 6;
    const int ttile    = (Wb >> 8) & 15;
    const int b        = Wb >> 12;

    // cooperative masked A-tile build: 8192 bf16, 64 per thread, coalesced
    const float* xbase = x  + ((size_t)(b * T_ + ttile * 16)) * D_;
    const float* zrow  = zx + ((size_t)(g * B_ + b)) * D_;
#pragma unroll
    for (int it = 0; it < 8; ++it) {
        int idx = it * 1024 + tid * 8;                 // 2 rows per pass
        int row = idx >> 9;
        int k   = idx & 511;
        const float* xp = xbase + (size_t)row * D_ + k;
        const float* zp = zrow + k;
        v8bf tmp;
#pragma unroll
        for (int e = 0; e < 8; ++e) tmp[e] = (bf16_t)(xp[e] * zp[e]);
        *(v8bf*)&sA[idx] = tmp;
    }
    __syncthreads();

    const int lane    = tid & 31;
    const int warp    = tid >> 5;
    const int coltile = coltile0 + warp;
    const int colbase = coltile << 4;
    const int m  = lane & 15;
    const int kg = lane >> 4;

    const bf16_t* bcol = KxT + (size_t)(colbase + m) * D_ + kg * 16;

    v8f acc;
    const float bv = bias[colbase + m];                // same for all 8 rows
#pragma unroll
    for (int r = 0; r < 8; ++r) acc[r] = bv;

#pragma unroll 4
    for (int kk = 0; kk < 16; ++kk) {
        const int kbase = kk * 32;
        // A fragment from LDS: one contiguous 16-element run (2x ds_load_b128)
        const bf16_t* ap = sA + m * 512 + kbase + kg * 16;
        v8bf lo = *(const v8bf*)(ap);
        v8bf hi = *(const v8bf*)(ap + 8);
        v16bf afrag = __builtin_shufflevector(lo, hi,
                          0,1,2,3,4,5,6,7,8,9,10,11,12,13,14,15);
        // B fragment: 16 contiguous K per lane (weights are column-major)
        v16bf bfrag = *(const v16bf*)(bcol + kbase);
        acc = __builtin_amdgcn_wmma_f32_16x16x32_bf16(false, afrag, false, bfrag,
                                                      (short)0, acc, false, false);
    }
    // C layout: VGPR r -> M = r + 8*kg ; N = lane&15
    float* outp = XM + (((size_t)(ttile * 16 + kg * 8) * B_ + b) * G4) + colbase + m;
#pragma unroll
    for (int r = 0; r < 8; ++r) outp[(size_t)r * (B_ * G4)] = acc[r];
}

// ---------------------------------------------------------------------------
// Per-step recurrent GEMM: Mt(64 x 4096) += hm[g](64 x 1024) @ KhT.
// One wave per 16-column tile; a block's 4 waves share gate g, so the 64x32
// bf16 A chunk is staged once per block into LDS with async global->LDS
// copies, double-buffered so chunk k+1 copies overlap chunk k's WMMAs.
// KhT streams from L2 exactly once per step (8.4 MB, the roofline minimum).
// ---------------------------------------------------------------------------
__global__ __launch_bounds__(128)
void recur_gemm_kernel(const bf16_t* __restrict__ hm, const bf16_t* __restrict__ KhT,
                       float* __restrict__ Mt) {
    __shared__ bf16_t sA[2][64 * 32];                  // 2 x 4 KB double buffer

    const int tid     = threadIdx.x;
    const int lane    = tid & 31;
    const int warp    = tid >> 5;
    const int coltile = blockIdx.x * 4 + warp;         // 0..255
    const int g       = blockIdx.x >> 4;               // uniform per block
    const int colbase = coltile << 4;
    const int m  = lane & 15;
    const int kg = lane >> 4;

    const bf16_t* abase = hm + (size_t)(g * B_) * H_;
    const bf16_t* bcol  = KhT + (size_t)(colbase + m) * H_ + kg * 16;

    // cooperative async stage of one 64x32 A chunk (256 x 16B transfers)
    auto stage = [&](int buf, int kbase) {
#pragma unroll
        for (int it = 0; it < 2; ++it) {
            int item = tid + it * 128;                 // 0..255
            int row  = item >> 2;
            int sub  = item & 3;
            async_copy16(abase + (size_t)row * H_ + kbase + sub * 8,
                         &sA[buf][row * 32 + sub * 8]);
        }
    };

    v8f acc[4];
#pragma unroll
    for (int rt = 0; rt < 4; ++rt) {
        const float* mp = Mt + (size_t)(rt * 16 + kg * 8) * G4 + colbase + m;
#pragma unroll
        for (int r = 0; r < 8; ++r) acc[rt][r] = mp[(size_t)r * G4];
    }

    stage(0, 0);
    for (int kk = 0; kk < 32; ++kk) {
        async_wait_all();                              // own async copies landed
        __syncthreads();                               // everyone's copies visible
        if (kk < 31) stage((kk + 1) & 1, (kk + 1) * 32);

        const int kbase = kk * 32;
        v16bf bfrag = *(const v16bf*)(bcol + kbase);
        const bf16_t* as = sA[kk & 1];
#pragma unroll
        for (int rt = 0; rt < 4; ++rt) {
            const bf16_t* ap = as + (rt * 16 + m) * 32 + kg * 16;
            v8bf lo = *(const v8bf*)(ap);
            v8bf hi = *(const v8bf*)(ap + 8);
            v16bf afrag = __builtin_shufflevector(lo, hi,
                              0,1,2,3,4,5,6,7,8,9,10,11,12,13,14,15);
            acc[rt] = __builtin_amdgcn_wmma_f32_16x16x32_bf16(false, afrag, false, bfrag,
                                                              (short)0, acc[rt], false, false);
        }
    }
#pragma unroll
    for (int rt = 0; rt < 4; ++rt) {
        float* mp = Mt + (size_t)(rt * 16 + kg * 8) * G4 + colbase + m;
#pragma unroll
        for (int r = 0; r < 8; ++r) mp[(size_t)r * G4] = acc[rt][r];
    }
}

// ---------------------------------------------------------------------------
// Per-step pointwise: gates, cell/hidden update, masked bf16 h for next step.
// Gate order (reference): i=M[0], f=M[1], o=M[2], g=M[3].
// ---------------------------------------------------------------------------
__global__ void act_kernel(const float* __restrict__ Mt, const float* __restrict__ zh,
                           float* __restrict__ c, bf16_t* __restrict__ hm,
                           float* __restrict__ out, int t) {
    int idx = blockIdx.x * 256 + threadIdx.x;          // 0 .. B*H-1
    int b = idx >> 10;
    int n = idx & (H_ - 1);
    const float* mrow = Mt + (size_t)b * G4;
    float mi = mrow[n], mf = mrow[H_ + n], mo = mrow[2 * H_ + n], mg = mrow[3 * H_ + n];
    float i_ = 1.0f / (1.0f + __expf(-mi));
    float f_ = 1.0f / (1.0f + __expf(-mf));
    float o_ = 1.0f / (1.0f + __expf(-mo));
    float g_ = tanhf(mg);
    float cn = f_ * c[idx] + i_ * g_;
    c[idx] = cn;
    float h = o_ * tanhf(cn);
    out[((size_t)b * T_ + t) * H_ + n] = h;
#pragma unroll
    for (int q = 0; q < 4; ++q) {
        int j = (q * B_ + b) * H_ + n;
        hm[j] = (bf16_t)(h * zh[j]);
    }
}

// ---------------------------------------------------------------------------
extern "C" void kernel_launch(void* const* d_in, const int* in_sizes, int n_in,
                              void* d_out, int out_size, void* d_ws, size_t ws_size,
                              hipStream_t stream) {
    (void)in_sizes; (void)n_in; (void)out_size; (void)ws_size;
    const float* inputs  = (const float*)d_in[0];   // (B,T,D)
    const float* kern    = (const float*)d_in[1];   // (D,4H)
    const float* rkern   = (const float*)d_in[2];   // (H,4H)
    const float* bias    = (const float*)d_in[3];   // (4H)
    const float* dropout = (const float*)d_in[4];   // (1)
    const float* ux      = (const float*)d_in[5];   // (4,B,D)
    const float* uh      = (const float*)d_in[6];   // (4,B,H)
    float* out = (float*)d_out;                     // (B,T,H)

    // workspace carve-up (all offsets 256B-aligned by construction)
    char* w = (char*)d_ws;
    float*  zx  = (float*)w;  w += (size_t)4 * B_ * D_ * 4;   // 0.5 MB
    float*  zh  = (float*)w;  w += (size_t)4 * B_ * H_ * 4;   // 1 MB
    float*  cst = (float*)w;  w += (size_t)B_ * H_ * 4;       // 256 KB cell state
    bf16_t* hm  = (bf16_t*)w; w += (size_t)4 * B_ * H_ * 2;   // 512 KB masked h
    bf16_t* KxT = (bf16_t*)w; w += (size_t)G4 * D_ * 2;       // 4 MB
    bf16_t* KhT = (bf16_t*)w; w += (size_t)G4 * H_ * 2;       // 8 MB
    float*  XM  = (float*)w;                                  // 256 MB gate staging

    mask_kernel<<<(4 * B_ * D_ + 4 * B_ * H_) / 256, 256, 0, stream>>>(ux, uh, dropout, zx, zh);
    cvt_kernel<<<(G4 * D_ + G4 * H_) / 256, 256, 0, stream>>>(kern, rkern, KxT, KhT);
    init_kernel<<<(4 * B_ * H_) / 256, 256, 0, stream>>>(cst, hm);

    // 64*16*256 tiles, 4 waves per block
    input_proj_kernel<<<(B_ * 16 * 256) / 4, 128, 0, stream>>>(inputs, zx, KxT, bias, XM);

    for (int t = 0; t < T_; ++t) {
        float* Mt = XM + (size_t)t * B_ * G4;
        recur_gemm_kernel<<<64, 128, 0, stream>>>(hm, KhT, Mt);
        act_kernel<<<(B_ * H_) / 256, 256, 0, stream>>>(Mt, zh, cst, hm, out, t);
    }
}
